// EncoderBlock_69191923138685
// MI455X (gfx1250) — matrix operand order repaired and checked by reading, not compiled, not run on hardware
//
#include <hip/hip_runtime.h>

// ---------------------------------------------------------------------------
// Types for CDNA5 (gfx1250) wave32 WMMA
// ---------------------------------------------------------------------------
typedef __bf16 bf16;
typedef __attribute__((ext_vector_type(16))) __bf16 v16bf;
typedef __attribute__((ext_vector_type(8)))  __bf16 bf16x8;
typedef __attribute__((ext_vector_type(8)))  float  v8f;

union FragBF { v16bf v; bf16x8 h[2]; };

// Problem dims (fixed by the reference)
constexpr int B_   = 2;
constexpr int S_   = 2048;
constexpr int D_   = 1024;
constexpr int H_   = 16;
constexpr int DFF_ = 4096;
constexpr int DK_  = 64;
constexpr int M_   = B_ * S_;   // 4096 token rows

// ---------------------------------------------------------------------------
// Tiled transpose + fp32->bf16 convert:  out[c][r] = (bf16)in[r][c]
// in: fp32 [R][C], out: bf16 [C][R].  Grid: (C/32, R/32), 256 threads.
// Both global read and write are coalesced; only the tiny staging tile uses
// scalar LDS stores (keeps them out of the GEMM hot loop).
// ---------------------------------------------------------------------------
__global__ __launch_bounds__(256) void cvt_transpose_kernel(
    const float* __restrict__ in, bf16* __restrict__ out, int R, int C)
{
    __shared__ bf16 tile[32][33];
    const int c0 = blockIdx.x * 32, r0 = blockIdx.y * 32;
    const int tx = threadIdx.x & 31, ty = threadIdx.x >> 5;   // 32 x 8
    #pragma unroll
    for (int i = 0; i < 4; ++i) {
        const int r = ty + i * 8;
        tile[r][tx] = (bf16)in[(size_t)(r0 + r) * C + c0 + tx];
    }
    __syncthreads();
    #pragma unroll
    for (int i = 0; i < 4; ++i) {
        const int c = ty + i * 8;
        out[(size_t)(c0 + c) * R + r0 + tx] = tile[tx][c];
    }
}

// ---------------------------------------------------------------------------
// Per-head V transpose (bf16 -> bf16): Vt[b][h][dk][s] = V[b][s][h][dk]
// Grid: (S/32, DK/32, B*H), 256 threads.
// ---------------------------------------------------------------------------
__global__ __launch_bounds__(256) void transpose_v_kernel(
    const bf16* __restrict__ in, bf16* __restrict__ out)
{
    __shared__ bf16 tile[32][33];
    const int bh = blockIdx.z;                 // b*H + h
    const int b = bh / H_, hd = bh % H_;
    const int s0 = blockIdx.x * 32, d0 = blockIdx.y * 32;
    const int tx = threadIdx.x & 31, ty = threadIdx.x >> 5;
    #pragma unroll
    for (int i = 0; i < 4; ++i) {
        const int s = ty + i * 8;
        tile[s][tx] = in[(size_t)(b * S_ + s0 + s) * D_ + hd * DK_ + d0 + tx];
    }
    __syncthreads();
    #pragma unroll
    for (int i = 0; i < 4; ++i) {
        const int d = ty + i * 8;
        out[((size_t)bh * DK_ + d0 + d) * S_ + s0 + tx] = tile[tx][d];
    }
}

// ---------------------------------------------------------------------------
// LayerNorm (torch semantics: unbiased std, alpha*(x-mean)/(std+eps)+beta)
// One 256-thread block (8 wave32) per token row of D_=1024.
// ---------------------------------------------------------------------------
__global__ __launch_bounds__(256) void ln_bf16_kernel(
    const float* __restrict__ x, const float* __restrict__ alpha,
    const float* __restrict__ beta, bf16* __restrict__ out)
{
    const int row = blockIdx.x;
    const float* xr = x + (size_t)row * D_;
    float s = 0.f, s2 = 0.f;
    for (int i = threadIdx.x; i < D_; i += 256) {
        float v = xr[i];
        s += v; s2 += v * v;
    }
    __shared__ float red[2][8];
    #pragma unroll
    for (int o = 16; o > 0; o >>= 1) {
        s  += __shfl_down(s,  o, 32);
        s2 += __shfl_down(s2, o, 32);
    }
    const int wave = threadIdx.x >> 5, lane = threadIdx.x & 31;
    if (lane == 0) { red[0][wave] = s; red[1][wave] = s2; }
    __syncthreads();
    s = 0.f; s2 = 0.f;
    #pragma unroll
    for (int w = 0; w < 8; ++w) { s += red[0][w]; s2 += red[1][w]; }

    const float mean  = s * (1.f / (float)D_);
    const float var   = fmaxf((s2 - (float)D_ * mean * mean) / (float)(D_ - 1), 0.f);
    const float scale = alpha[0] / (sqrtf(var) + 1e-6f);
    const float shift = beta[0];
    for (int i = threadIdx.x; i < D_; i += 256)
        out[(size_t)row * D_ + i] = (bf16)((xr[i] - mean) * scale + shift);
}

// ---------------------------------------------------------------------------
// Tiled bf16 GEMM via v_wmma_f32_16x16x32_bf16.
//   out = act(A[M,K] @ W[K,N] + bias[N]) (+ resid[M,N] fp32)
// W is supplied PRE-TRANSPOSED as WtT[N][K] so both LDS stages are pure
// 16-byte ds_store_b128, and both fragment reads are 16-byte ds_load_b128
// matching the CDNA5 16-bit A/B VGPR layouts.
// Block tile 128x128x32, 8 waves (4 M-groups x 2 N-groups), wave = 32x64.
// Software-pipelined: next K-tile global loads issue before WMMA consumption.
// ---------------------------------------------------------------------------
template<bool RELU, bool RESIDUAL>
__global__ __launch_bounds__(256, 2) void gemm_bf16_wmma(
    const bf16* __restrict__ A, const bf16* __restrict__ WtT,
    const float* __restrict__ bias, const float* __restrict__ resid,
    float* __restrict__ outF, bf16* __restrict__ outB,
    int Ndim, int Kdim)
{
    constexpr int BM = 128, BN = 128, BK = 32;
    constexpr int LDT = BK + 16;   // 48 elems = 96 B row stride (16B multiple)
    __shared__ bf16 As[BM][LDT];   // [m][k]
    __shared__ bf16 Bs[BN][LDT];   // [n][k]

    const int tid  = threadIdx.x;
    const int wave = tid >> 5, lane = tid & 31;
    const int hgrp = lane >> 4, l16 = lane & 15;
    const int wm = wave >> 1, wn = wave & 1;
    const int blockM = blockIdx.y * BM;
    const int blockN = blockIdx.x * BN;

    // staging indices: each thread owns rows lr and lr+64, 8 contiguous k
    const int lr = tid >> 2;             // 0..63
    const int lc = (tid & 3) * 8;        // 0,8,16,24

    v8f acc[2][4];
    #pragma unroll
    for (int ms = 0; ms < 2; ++ms)
        #pragma unroll
        for (int ns = 0; ns < 4; ++ns)
            #pragma unroll
            for (int i = 0; i < 8; ++i) acc[ms][ns][i] = 0.f;

    const int nk = Kdim / BK;
    bf16x8 ra[2], rb[2];

    // preload K-tile 0 into registers
    #pragma unroll
    for (int c = 0; c < 2; ++c) {
        ra[c] = *(const bf16x8*)(A   + (size_t)(blockM + c * 64 + lr) * Kdim + lc);
        rb[c] = *(const bf16x8*)(WtT + (size_t)(blockN + c * 64 + lr) * Kdim + lc);
    }

    for (int kt = 0; kt < nk; ++kt) {
        // stage current tile (pure b128 DS stores)
        #pragma unroll
        for (int c = 0; c < 2; ++c) {
            *(bf16x8*)&As[c * 64 + lr][lc] = ra[c];
            *(bf16x8*)&Bs[c * 64 + lr][lc] = rb[c];
        }
        __syncthreads();

        // issue next tile's global loads before consuming LDS
        if (kt + 1 < nk) {
            const int k0 = (kt + 1) * BK;
            #pragma unroll
            for (int c = 0; c < 2; ++c) {
                ra[c] = *(const bf16x8*)(A   + (size_t)(blockM + c * 64 + lr) * Kdim + k0 + lc);
                rb[c] = *(const bf16x8*)(WtT + (size_t)(blockN + c * 64 + lr) * Kdim + k0 + lc);
            }
        }

        // fragments (contiguous 16B ds_load_b128 per ISA layouts)
        FragBF aF[2], bF[4];
        #pragma unroll
        for (int ms = 0; ms < 2; ++ms) {
            // A 16x32: lanes0-15 K=[0..7|16..23], lanes16-31 K=[8..15|24..31]
            const bf16* src = &As[wm * 32 + ms * 16 + l16][0];
            aF[ms].h[0] = *(const bf16x8*)(src + hgrp * 8);
            aF[ms].h[1] = *(const bf16x8*)(src + 16 + hgrp * 8);
        }
        #pragma unroll
        for (int ns = 0; ns < 4; ++ns) {
            // B 32x16: lanes0-15 K=[0..7|16..23], lanes16-31 K=[8..15|24..31] (col = l16)
            const bf16* src = &Bs[wn * 64 + ns * 16 + l16][hgrp * 8];
            bF[ns].h[0] = *(const bf16x8*)(src);
            bF[ns].h[1] = *(const bf16x8*)(src + 16);
        }
        #pragma unroll
        for (int ms = 0; ms < 2; ++ms)
            #pragma unroll
            for (int ns = 0; ns < 4; ++ns)
                acc[ms][ns] = __builtin_amdgcn_wmma_f32_16x16x32_bf16(
                    false, aF[ms].v, false, bF[ns].v, (short)0, acc[ms][ns],
                    false, false);
        __syncthreads();
    }

    // epilogue: bias (+ReLU) (+fp32 residual)
    #pragma unroll
    for (int ms = 0; ms < 2; ++ms) {
        #pragma unroll
        for (int ns = 0; ns < 4; ++ns) {
            const int n  = blockN + wn * 64 + ns * 16 + l16;
            const float bv = bias[n];
            #pragma unroll
            for (int r = 0; r < 8; ++r) {
                const int mrow = blockM + wm * 32 + ms * 16 + hgrp * 8 + r;
                float val = acc[ms][ns][r] + bv;
                if (RELU) val = fmaxf(val, 0.f);
                const size_t idx = (size_t)mrow * Ndim + n;
                if (RESIDUAL) outF[idx] = val + resid[idx];
                else          outB[idx] = (bf16)val;
            }
        }
    }
}

// ---------------------------------------------------------------------------
// Flash attention: per block = (128 Q rows, one head, one batch).
// 8 waves; each wave owns a 16-row Q tile. 64-key chunks streamed through LDS:
//   Ks[key][dk]  from K  (row-major),     pure b128 stage
//   Vs[dk][key]  from Vt (pre-transposed), pure b128 stage
// P goes C-layout -> per-wave LDS -> A-layout reload. K/V chunk loads are
// software-pipelined ahead of the WMMA/softmax work.
// ---------------------------------------------------------------------------
__global__ __launch_bounds__(256, 2) void attn_kernel(
    const bf16* __restrict__ Q, const bf16* __restrict__ K,
    const bf16* __restrict__ Vt, const int* __restrict__ mask,
    bf16* __restrict__ ctx)
{
    constexpr int TQ = 128, TK = 64, LDK = DK_ + 8;   // 72-elem rows (144 B)
    __shared__ bf16 Ks[TK][LDK];
    __shared__ bf16 Vs[DK_][TK + 8];
    __shared__ bf16 Ps[8][16][TK + 8];   // per-wave P staging

    const int tid  = threadIdx.x;
    const int wave = tid >> 5, lane = tid & 31;
    const int hgrp = lane >> 4, l16 = lane & 15;
    const int b  = blockIdx.z, hd = blockIdx.y;
    const int bh = b * H_ + hd;
    const int q0 = blockIdx.x * TQ;

    // staging indices: K rows (key), V rows (dk); 8 contiguous elems each
    const int srow = tid >> 3;           // 0..31
    const int sc8  = (tid & 7) * 8;      // 0..56

    // Q A-fragments for this wave's 16 rows (K-dim = DK_=64 -> 2 fragments)
    const size_t qoff = (size_t)(b * S_ + q0 + wave * 16 + l16) * D_ + hd * DK_;
    FragBF aQ[2];
    #pragma unroll
    for (int kk = 0; kk < 2; ++kk) {
        aQ[kk].h[0] = *(const bf16x8*)(Q + qoff + kk * 32 + hgrp * 8);
        aQ[kk].h[1] = *(const bf16x8*)(Q + qoff + kk * 32 + 16 + hgrp * 8);
    }

    float m_run[8], l_run[8];
    v8f accO[4];
    #pragma unroll
    for (int r = 0; r < 8; ++r) { m_run[r] = -3.0e38f; l_run[r] = 0.f; }
    #pragma unroll
    for (int ns = 0; ns < 4; ++ns)
        #pragma unroll
        for (int r = 0; r < 8; ++r) accO[ns][r] = 0.f;

    bf16x8 rk[2], rv[2];
    // preload chunk 0
    #pragma unroll
    for (int c = 0; c < 2; ++c) {
        rk[c] = *(const bf16x8*)(K  + (size_t)(b * S_ + c * 32 + srow) * D_ + hd * DK_ + sc8);
        rv[c] = *(const bf16x8*)(Vt + ((size_t)bh * DK_ + c * 32 + srow) * S_ + sc8);
    }

    for (int kc = 0; kc < S_; kc += TK) {
        // stage current K/V chunk (pure b128 DS stores)
        #pragma unroll
        for (int c = 0; c < 2; ++c) {
            *(bf16x8*)&Ks[c * 32 + srow][sc8] = rk[c];
            *(bf16x8*)&Vs[c * 32 + srow][sc8] = rv[c];
        }
        __syncthreads();

        // prefetch next chunk into registers
        if (kc + TK < S_) {
            #pragma unroll
            for (int c = 0; c < 2; ++c) {
                rk[c] = *(const bf16x8*)(K  + (size_t)(b * S_ + kc + TK + c * 32 + srow) * D_ + hd * DK_ + sc8);
                rv[c] = *(const bf16x8*)(Vt + ((size_t)bh * DK_ + c * 32 + srow) * S_ + kc + TK + sc8);
            }
        }

        // --- scores: S = Q @ K^T * 1/sqrt(DK), with mask -------------------
        v8f sc[4];
        #pragma unroll
        for (int j = 0; j < 4; ++j) {
            #pragma unroll
            for (int r = 0; r < 8; ++r) sc[j][r] = 0.f;
            #pragma unroll
            for (int kk = 0; kk < 2; ++kk) {
                FragBF bK;
                const bf16* src = &Ks[j * 16 + l16][kk * 32 + hgrp * 8];
                bK.h[0] = *(const bf16x8*)(src);
                bK.h[1] = *(const bf16x8*)(src + 16);
                sc[j] = __builtin_amdgcn_wmma_f32_16x16x32_bf16(
                    false, aQ[kk].v, false, bK.v, (short)0, sc[j], false, false);
            }
            const bool masked = (mask[b * S_ + kc + j * 16 + l16] == 0);
            #pragma unroll
            for (int r = 0; r < 8; ++r)
                sc[j][r] = masked ? -1.0e9f : sc[j][r] * 0.125f;
        }

        // --- online softmax (row reductions across the 16-lane half-group) -
        #pragma unroll
        for (int r = 0; r < 8; ++r) {
            float mx = fmaxf(fmaxf(sc[0][r], sc[1][r]), fmaxf(sc[2][r], sc[3][r]));
            #pragma unroll
            for (int off = 8; off > 0; off >>= 1)
                mx = fmaxf(mx, __shfl_xor(mx, off, 32));
            mx = fmaxf(mx, m_run[r]);
            const float corr = __expf(m_run[r] - mx);
            m_run[r] = mx;
            float rs = 0.f;
            #pragma unroll
            for (int j = 0; j < 4; ++j) {
                const float p = __expf(sc[j][r] - mx);
                sc[j][r] = p;
                rs += p;
            }
            #pragma unroll
            for (int off = 8; off > 0; off >>= 1)
                rs += __shfl_xor(rs, off, 32);
            l_run[r] = l_run[r] * corr + rs;
            #pragma unroll
            for (int ns = 0; ns < 4; ++ns) accO[ns][r] *= corr;
            // C-layout row = r + 8*hgrp, col = j*16 + l16
            #pragma unroll
            for (int j = 0; j < 4; ++j)
                Ps[wave][hgrp * 8 + r][j * 16 + l16] = (bf16)sc[j][r];
        }

        // --- P (A-layout reload) @ V ---------------------------------------
        FragBF aP[2];
        #pragma unroll
        for (int kk = 0; kk < 2; ++kk) {
            const bf16* src = &Ps[wave][l16][kk * 32];
            aP[kk].h[0] = *(const bf16x8*)(src + hgrp * 8);
            aP[kk].h[1] = *(const bf16x8*)(src + 16 + hgrp * 8);
        }
        #pragma unroll
        for (int ns = 0; ns < 4; ++ns) {
            #pragma unroll
            for (int kk = 0; kk < 2; ++kk) {
                FragBF bV;
                const bf16* src = &Vs[ns * 16 + l16][kk * 32 + hgrp * 8];
                bV.h[0] = *(const bf16x8*)(src);
                bV.h[1] = *(const bf16x8*)(src + 16);
                accO[ns] = __builtin_amdgcn_wmma_f32_16x16x32_bf16(
                    false, aP[kk].v, false, bV.v, (short)0, accO[ns], false, false);
            }
        }
        __syncthreads();
    }

    // --- normalize and write ctx (bf16) ------------------------------------
    #pragma unroll
    for (int ns = 0; ns < 4; ++ns) {
        #pragma unroll
        for (int r = 0; r < 8; ++r) {
            const int row = q0 + wave * 16 + hgrp * 8 + r;
            const float o = accO[ns][r] / l_run[r];
            ctx[(size_t)(b * S_ + row) * D_ + hd * DK_ + ns * 16 + l16] = (bf16)o;
        }
    }
}

// ---------------------------------------------------------------------------
// Host launcher
// ---------------------------------------------------------------------------
extern "C" void kernel_launch(void* const* d_in, const int* in_sizes, int n_in,
                              void* d_out, int out_size, void* d_ws, size_t ws_size,
                              hipStream_t stream)
{
    (void)in_sizes; (void)n_in; (void)out_size; (void)ws_size;

    const float* x    = (const float*)d_in[0];
    const int*   mask = (const int*)  d_in[1];
    const float* Wq = (const float*)d_in[2];  const float* bq = (const float*)d_in[3];
    const float* Wk = (const float*)d_in[4];  const float* bk = (const float*)d_in[5];
    const float* Wv = (const float*)d_in[6];  const float* bv = (const float*)d_in[7];
    const float* Wo = (const float*)d_in[8];  const float* bo = (const float*)d_in[9];
    const float* W1 = (const float*)d_in[10]; const float* b1 = (const float*)d_in[11];
    const float* W2 = (const float*)d_in[12]; const float* b2 = (const float*)d_in[13];
    const float* alpha1 = (const float*)d_in[14]; const float* beta1 = (const float*)d_in[15];
    const float* alpha2 = (const float*)d_in[16]; const float* beta2 = (const float*)d_in[17];
    float* out = (float*)d_out;

    // Carve workspace (~128 MiB)
    char* ws = (char*)d_ws;
    auto carve = [&](size_t bytes) -> char* {
        char* p = ws;
        ws += (bytes + 255) & ~(size_t)255;
        return p;
    };
    const size_t tokBF = (size_t)M_ * D_ * sizeof(bf16);
    bf16*  xn1  = (bf16*) carve(tokBF);
    bf16*  qb   = (bf16*) carve(tokBF);
    bf16*  kb   = (bf16*) carve(tokBF);
    bf16*  vb   = (bf16*) carve(tokBF);
    bf16*  vtb  = (bf16*) carve(tokBF);          // V transposed per head
    bf16*  ctx  = (bf16*) carve(tokBF);
    float* x2   = (float*)carve((size_t)M_ * D_ * sizeof(float));
    bf16*  xn2  = (bf16*) carve(tokBF);
    bf16*  hbuf = (bf16*) carve((size_t)M_ * DFF_ * sizeof(bf16));
    bf16*  wqT  = (bf16*) carve((size_t)D_ * D_ * sizeof(bf16));     // [N][K]
    bf16*  wkT  = (bf16*) carve((size_t)D_ * D_ * sizeof(bf16));
    bf16*  wvT  = (bf16*) carve((size_t)D_ * D_ * sizeof(bf16));
    bf16*  woT  = (bf16*) carve((size_t)D_ * D_ * sizeof(bf16));
    bf16*  w1T  = (bf16*) carve((size_t)D_ * DFF_ * sizeof(bf16));   // [DFF][D]
    bf16*  w2T  = (bf16*) carve((size_t)DFF_ * D_ * sizeof(bf16));   // [D][DFF]

    // Weight convert+transpose (fp32 [K][N] -> bf16 [N][K]); every call.
    const dim3 tDD(D_ / 32, D_ / 32);
    cvt_transpose_kernel<<<tDD, 256, 0, stream>>>(Wq, wqT, D_, D_);
    cvt_transpose_kernel<<<tDD, 256, 0, stream>>>(Wk, wkT, D_, D_);
    cvt_transpose_kernel<<<tDD, 256, 0, stream>>>(Wv, wvT, D_, D_);
    cvt_transpose_kernel<<<tDD, 256, 0, stream>>>(Wo, woT, D_, D_);
    cvt_transpose_kernel<<<dim3(DFF_ / 32, D_ / 32),  256, 0, stream>>>(W1, w1T, D_, DFF_);
    cvt_transpose_kernel<<<dim3(D_ / 32,  DFF_ / 32), 256, 0, stream>>>(W2, w2T, DFF_, D_);

    // 1) pre-norm 1
    ln_bf16_kernel<<<M_, 256, 0, stream>>>(x, alpha1, beta1, xn1);

    // 2) Q/K/V projections
    const dim3 gDD(D_ / 128, M_ / 128);
    gemm_bf16_wmma<false, false><<<gDD, 256, 0, stream>>>(
        xn1, wqT, bq, nullptr, nullptr, qb, D_, D_);
    gemm_bf16_wmma<false, false><<<gDD, 256, 0, stream>>>(
        xn1, wkT, bk, nullptr, nullptr, kb, D_, D_);
    gemm_bf16_wmma<false, false><<<gDD, 256, 0, stream>>>(
        xn1, wvT, bv, nullptr, nullptr, vb, D_, D_);

    // 2b) per-head V transpose for b128-only attention staging
    transpose_v_kernel<<<dim3(S_ / 32, DK_ / 32, B_ * H_), 256, 0, stream>>>(vb, vtb);

    // 3) flash attention
    attn_kernel<<<dim3(S_ / 128, H_, B_), 256, 0, stream>>>(qb, kb, vtb, mask, ctx);

    // 4) output projection + residual (fp32)
    gemm_bf16_wmma<false, true><<<gDD, 256, 0, stream>>>(
        ctx, woT, bo, x, x2, nullptr, D_, D_);

    // 5) pre-norm 2
    ln_bf16_kernel<<<M_, 256, 0, stream>>>(x2, alpha2, beta2, xn2);

    // 6) FFN up (ReLU fused)
    gemm_bf16_wmma<true, false><<<dim3(DFF_ / 128, M_ / 128), 256, 0, stream>>>(
        xn2, w1T, b1, nullptr, nullptr, hbuf, DFF_, D_);

    // 7) FFN down + residual -> final output (fp32)
    gemm_bf16_wmma<false, true><<<gDD, 256, 0, stream>>>(
        hbuf, w2T, b2, x2, out, nullptr, D_, DFF_);
}
